// LearnableHash_23347442221328
// MI455X (gfx1250) — compile-verified, matrix operand order checked by reading, not run on hardware
//
#include <hip/hip_runtime.h>
#include <hip/hip_bf16.h>
#include <math.h>

#define RADIUS_F  1.3f
#define STEP_F    0.0176f
#define B_RAYS    2048
#define NSAMP     256
#define M_PTS     (B_RAYS * NSAMP)

typedef __attribute__((ext_vector_type(16))) _Float16 v16h;
typedef __attribute__((ext_vector_type(8)))  float    v8f;

// ---------------- workspace layout (bytes) ----------------
#define WS_DIRN   0                          // 2048 * float4
#define WS_TNEAR  (WS_DIRN  + 2048*16)       // 2048 * f32
#define WS_TMAX   (WS_TNEAR + 2048*4)
#define WS_HIT    (WS_TMAX  + 2048*4)        // 2048 * i32
#define WS_FRAG   (WS_HIT   + 2048*4)        // 20 frags * 32 lanes * 16 halfs
#define WS_FT     (WS_FRAG  + 20*32*16*2)    // 32768 voxels * 32 f32 (channel-contig)
#define WS_G1T    (WS_FT    + 32768*32*4)    // 128^3 voxels * float4
#define WS_SIGMA  (WS_G1T   + 2097152*16)    // M_PTS f32
#define WS_RGB    (WS_SIGMA + M_PTS*4)       // M_PTS * 3 f32

// ---- K index map for f16 A/B fragments (16x16x32), from ISA 7.12.2 ----
__device__ __forceinline__ int kmap(int k, int lane, int h) {
  int lo = (lane & 31) < 16;
  int off = (h < 8) ? (lo ? h : h + 8) : (lo ? h + 8 : h + 16);
  return 32 * k + off;
}

// load A/B style fragment from a 64-half row (16B-aligned): two contiguous
// 8-half groups at [32k + (hi?8:0)] and [32k + (hi?8:0) + 16]
__device__ __forceinline__ v16h afrag(const _Float16* row, int k, int hi) {
  union { v16h v; float4 f[2]; } u;
  const float4* p = (const float4*)row;   // 8 float4 per row
  int i = 4 * k + hi;
  u.f[0] = p[i];
  u.f[1] = p[i + 2];
  return u.v;
}

// pre-swizzled B fragment: 32B contiguous per lane
__device__ __forceinline__ v16h bfrag(const _Float16* fb, int f, int lane) {
  union { v16h v; float4 f4[2]; } u;
  const float4* p = (const float4*)(fb + ((f << 5) + lane) * 16);
  u.f4[0] = p[0];
  u.f4[1] = p[1];
  return u.v;
}

#define WMMA_F16(C, A, Bm) \
  (C) = __builtin_amdgcn_wmma_f32_16x16x32_f16(false, (A), false, (Bm), (short)0, (C), false, false)

// ---------------- kernel 1: ray setup ----------------
__global__ void ray_setup_kernel(const float* __restrict__ rays_o,
                                 const float* __restrict__ rays_d,
                                 float4* __restrict__ dirn,
                                 float* __restrict__ tnear,
                                 float* __restrict__ tmaxo,
                                 int* __restrict__ hit) {
  int b = blockIdx.x * blockDim.x + threadIdx.x;
  if (b >= B_RAYS) return;
  float ox = rays_o[b*3+0], oy = rays_o[b*3+1], oz = rays_o[b*3+2];
  float dx = rays_d[b*3+0], dy = rays_d[b*3+1], dz = rays_d[b*3+2];
  float inv = rsqrtf(dx*dx + dy*dy + dz*dz);
  dx *= inv; dy *= inv; dz *= inv;
  const float eps = 1e-9f;
  float sx = (fabsf(dx) < eps) ? eps : dx;
  float sy = (fabsf(dy) < eps) ? eps : dy;
  float sz = (fabsf(dz) < eps) ? eps : dz;
  float ix = 1.f/sx, iy = 1.f/sy, iz = 1.f/sz;
  float t1x = (-RADIUS_F - ox)*ix, t2x = (RADIUS_F - ox)*ix;
  float t1y = (-RADIUS_F - oy)*iy, t2y = (RADIUS_F - oy)*iy;
  float t1z = (-RADIUS_F - oz)*iz, t2z = (RADIUS_F - oz)*iz;
  float tmin = fmaxf(fminf(t1x,t2x), fmaxf(fminf(t1y,t2y), fminf(t1z,t2z)));
  float tmx  = fminf(fmaxf(t1x,t2x), fminf(fmaxf(t1y,t2y), fmaxf(t1z,t2z)));
  float tn = fmaxf(tmin, 0.f);
  dirn[b]  = make_float4(dx, dy, dz, 0.f);
  tnear[b] = tn;
  tmaxo[b] = tmx;
  hit[b]   = (tmx > tn) ? 1 : 0;
}

// ---------------- kernel 2: transpose G1 [3][128^3] -> [128^3] float4 ----------------
__global__ void transpose_G1_kernel(const float* __restrict__ G1,
                                    float4* __restrict__ G1t) {
  int v = blockIdx.x * blockDim.x + threadIdx.x;
  if (v >= 2097152) return;
  G1t[v] = make_float4(G1[v], G1[v + 2097152], G1[v + 2*2097152], 0.f);
}

// ---------------- kernel 3: transpose F [32][32^3] -> [32^3][32] ----------------
__global__ void transpose_F_kernel(const float* __restrict__ F,
                                   float* __restrict__ Ft) {
  int v = blockIdx.x * blockDim.x + threadIdx.x;
  if (v >= 32768) return;
  #pragma unroll
  for (int c = 0; c < 32; ++c) Ft[v*32 + c] = F[c*32768 + v];
}

// ---------------- kernel 4: pre-swizzle weight fragments ----------------
// frag ids: 0..7 Wd1[n][k], 8..15 Wc1[n][k], 16..17 Wd2[k], 18..19 Wc2[k]
__global__ void pack_weights_kernel(const float* __restrict__ Wd1,
                                    const float* __restrict__ Wc1,
                                    const float* __restrict__ Wd2,
                                    const float* __restrict__ Wc2,
                                    _Float16* __restrict__ fb) {
  int tid = blockIdx.x * blockDim.x + threadIdx.x;
  if (tid >= 20*32*16) return;
  int h    = tid & 15;
  int lane = (tid >> 4) & 31;
  int f    = tid >> 9;
  int ncol = lane & 15;
  float val = 0.f;
  if (f < 8) {
    int n = f >> 1, k = f & 1;
    int K = kmap(k, lane, h);
    if (K < 32) val = Wd1[K*64 + n*16 + ncol];
  } else if (f < 16) {
    int g = f - 8; int n = g >> 1, k = g & 1;
    int K = kmap(k, lane, h);
    if (K < 35) val = Wc1[K*64 + n*16 + ncol];
  } else if (f < 18) {
    int K = kmap(f - 16, lane, h);
    if (ncol == 0) val = Wd2[K];
  } else {
    int K = kmap(f - 18, lane, h);
    if (ncol < 3) val = Wc2[K*3 + ncol];
  }
  fb[tid] = (_Float16)val;
}

// ---------------- trilinear helper ----------------
struct TriIdx { int x0,y0,z0,x1,y1,z1; float fx,fy,fz; };
__device__ __forceinline__ TriIdx tri_prep(float px, float py, float pz, int sz) {
  TriIdx r;
  float sm1 = (float)(sz - 1);
  float cx = fminf(fmaxf((px + 1.f) * 0.5f * sm1, 0.f), sm1);
  float cy = fminf(fmaxf((py + 1.f) * 0.5f * sm1, 0.f), sm1);
  float cz = fminf(fmaxf((pz + 1.f) * 0.5f * sm1, 0.f), sm1);
  r.x0 = (int)floorf(cx); r.y0 = (int)floorf(cy); r.z0 = (int)floorf(cz);
  r.fx = cx - (float)r.x0; r.fy = cy - (float)r.y0; r.fz = cz - (float)r.z0;
  r.x1 = (r.x0 + 1 < sz) ? r.x0 + 1 : sz - 1;
  r.y1 = (r.y0 + 1 < sz) ? r.y0 + 1 : sz - 1;
  r.z1 = (r.z0 + 1 < sz) ? r.z0 + 1 : sz - 1;
  return r;
}

// ---------------- kernel 5: fused sampling + WMMA MLPs ----------------
// block = 128 threads = 4 waves; wave handles 32 points (2 WMMA M-tiles)
__global__ void __launch_bounds__(128)
nerf_fused_kernel(const float*  __restrict__ rays_o,
                  const float4* __restrict__ dirn,
                  const float*  __restrict__ tnear,
                  const float4* __restrict__ G1t,
                  const float4* __restrict__ Ft4,
                  const _Float16* __restrict__ fb,
                  const float* __restrict__ bd1, const float* __restrict__ bd2,
                  const float* __restrict__ bc1, const float* __restrict__ bc2,
                  float* __restrict__ sigmaout, float* __restrict__ rgbout) {
  __shared__ __align__(16) _Float16 Xs[4][32][64];     // padded MLP inputs
  __shared__ __align__(16) _Float16 Hs[4][2][16][64];  // hd / hc staging

  int w    = threadIdx.x >> 5;
  int lane = threadIdx.x & 31;
  int m    = blockIdx.x * 128 + threadIdx.x;
  int b    = m >> 8;
  int s    = m & 255;

  // ---- point generation ----
  float4 d = dirn[b];
  float t  = tnear[b] + STEP_F * (float)s;
  float px = (rays_o[b*3+0] + t * d.x) * (1.f / RADIUS_F);
  float py = (rays_o[b*3+1] + t * d.y) * (1.f / RADIUS_F);
  float pz = (rays_o[b*3+2] + t * d.z) * (1.f / RADIUS_F);

  // ---- trilinear G1 (3 channels, float4 taps) ----
  TriIdx g = tri_prep(px, py, pz, 128);
  float g1x = 0.f, g1y = 0.f, g1z = 0.f;
  {
    float wx1 = g.fx, wx0 = 1.f - g.fx;
    float wy1 = g.fy, wy0 = 1.f - g.fy;
    float wz1 = g.fz, wz0 = 1.f - g.fz;
#define G1TAP(IX,IY,IZ,WT) { float4 v = G1t[(((IZ)*128 + (IY))*128 + (IX))]; \
    g1x += (WT)*v.x; g1y += (WT)*v.y; g1z += (WT)*v.z; }
    G1TAP(g.x0, g.y0, g.z0, wx0*wy0*wz0);
    G1TAP(g.x1, g.y0, g.z0, wx1*wy0*wz0);
    G1TAP(g.x0, g.y1, g.z0, wx0*wy1*wz0);
    G1TAP(g.x1, g.y1, g.z0, wx1*wy1*wz0);
    G1TAP(g.x0, g.y0, g.z1, wx0*wy0*wz1);
    G1TAP(g.x1, g.y0, g.z1, wx1*wy0*wz1);
    G1TAP(g.x0, g.y1, g.z1, wx0*wy1*wz1);
    G1TAP(g.x1, g.y1, g.z1, wx1*wy1*wz1);
#undef G1TAP
  }

  // ---- trilinear F (32 channels, channel-contiguous float4 taps) ----
  TriIdx q = tri_prep(g1x, g1y, g1z, 32);
  float fe[32];
  #pragma unroll
  for (int c = 0; c < 32; ++c) fe[c] = 0.f;
  {
    float wx1 = q.fx, wx0 = 1.f - q.fx;
    float wy1 = q.fy, wy0 = 1.f - q.fy;
    float wz1 = q.fz, wz0 = 1.f - q.fz;
#define FTAP(IX,IY,IZ,WT) { const float4* r = Ft4 + (size_t)(((IZ)*32 + (IY))*32 + (IX)) * 8; \
    float wt = (WT); \
    _Pragma("unroll") for (int j = 0; j < 8; ++j) { float4 v = r[j]; \
      fe[4*j+0] += wt*v.x; fe[4*j+1] += wt*v.y; fe[4*j+2] += wt*v.z; fe[4*j+3] += wt*v.w; } }
    FTAP(q.x0, q.y0, q.z0, wx0*wy0*wz0);
    FTAP(q.x1, q.y0, q.z0, wx1*wy0*wz0);
    FTAP(q.x0, q.y1, q.z0, wx0*wy1*wz0);
    FTAP(q.x1, q.y1, q.z0, wx1*wy1*wz0);
    FTAP(q.x0, q.y0, q.z1, wx0*wy0*wz1);
    FTAP(q.x1, q.y0, q.z1, wx1*wy0*wz1);
    FTAP(q.x0, q.y1, q.z1, wx0*wy1*wz1);
    FTAP(q.x1, q.y1, q.z1, wx1*wy1*wz1);
#undef FTAP
  }

  // ---- stage padded input row: [feats(32), dirn(3), zeros(29)] ----
  {
    _Float16* xr = &Xs[w][lane][0];
    #pragma unroll
    for (int c = 0; c < 32; ++c) xr[c] = (_Float16)fe[c];
    xr[32] = (_Float16)d.x; xr[33] = (_Float16)d.y; xr[34] = (_Float16)d.z;
    #pragma unroll
    for (int c = 35; c < 64; ++c) xr[c] = (_Float16)0.f;
  }
  __syncthreads();

  // ---- WMMA: 2 M-tiles of 16 points per wave ----
  int hi   = lane >> 4;
  int ncol = lane & 15;
  float bsig = bd2[0];

  for (int tt = 0; tt < 2; ++tt) {
    const _Float16* xrow = &Xs[w][tt*16 + ncol][0];
    v16h a0 = afrag(xrow, 0, hi);
    v16h a1 = afrag(xrow, 1, hi);

    // layer 1: density & color hidden (N = 64 -> 4 tiles, K = 64 -> 2 chunks)
    #pragma unroll
    for (int n = 0; n < 4; ++n) {
      v8f cd = {0.f,0.f,0.f,0.f,0.f,0.f,0.f,0.f};
      WMMA_F16(cd, a0, bfrag(fb, n*2 + 0, lane));
      WMMA_F16(cd, a1, bfrag(fb, n*2 + 1, lane));
      float bbd = bd1[n*16 + ncol];
      v8f cc = {0.f,0.f,0.f,0.f,0.f,0.f,0.f,0.f};
      WMMA_F16(cc, a0, bfrag(fb, 8 + n*2 + 0, lane));
      WMMA_F16(cc, a1, bfrag(fb, 8 + n*2 + 1, lane));
      float bbc = bc1[n*16 + ncol];
      #pragma unroll
      for (int v = 0; v < 8; ++v) {
        Hs[w][0][hi*8 + v][16*n + ncol] = (_Float16)fmaxf(cd[v] + bbd, 0.f);
        Hs[w][1][hi*8 + v][16*n + ncol] = (_Float16)fmaxf(cc[v] + bbc, 0.f);
      }
    }
    __syncthreads();

    // layer 2: sigma (col 0) and rgb (cols 0..2) via padded [64x16] weights
    const _Float16* hdrow = &Hs[w][0][ncol][0];
    const _Float16* hcrow = &Hs[w][1][ncol][0];
    v16h ad0 = afrag(hdrow, 0, hi), ad1 = afrag(hdrow, 1, hi);
    v16h ac0 = afrag(hcrow, 0, hi), ac1 = afrag(hcrow, 1, hi);
    v8f od = {0.f,0.f,0.f,0.f,0.f,0.f,0.f,0.f};
    WMMA_F16(od, ad0, bfrag(fb, 16, lane));
    WMMA_F16(od, ad1, bfrag(fb, 17, lane));
    v8f oc = {0.f,0.f,0.f,0.f,0.f,0.f,0.f,0.f};
    WMMA_F16(oc, ac0, bfrag(fb, 18, lane));
    WMMA_F16(oc, ac1, bfrag(fb, 19, lane));

    int mglob = blockIdx.x * 128 + w * 32 + tt * 16 + hi * 8;
    if (ncol == 0) {
      #pragma unroll
      for (int v = 0; v < 8; ++v) sigmaout[mglob + v] = od[v] + bsig;
    }
    if (ncol < 3) {
      float bsc = bc2[ncol];
      #pragma unroll
      for (int v = 0; v < 8; ++v) rgbout[(size_t)(mglob + v) * 3 + ncol] = oc[v] + bsc;
    }
    __syncthreads();
  }
}

// ---------------- kernel 6: per-ray volume rendering ----------------
__global__ void integrate_kernel(const float* __restrict__ sigma,
                                 const float* __restrict__ rgb,
                                 const float* __restrict__ tnear,
                                 const float* __restrict__ tmaxi,
                                 const int* __restrict__ hit,
                                 float* __restrict__ out) {
  int b = blockIdx.x * blockDim.x + threadIdx.x;
  if (b >= B_RAYS) return;
  float tn = tnear[b], tx = tmaxi[b];
  int h = hit[b];
  float T = 1.f, acc = 0.f, o0 = 0.f, o1 = 0.f, o2 = 0.f;
  for (int s = 0; s < NSAMP; ++s) {
    int m = b * NSAMP + s;
    float t = tn + STEP_F * (float)s;
    bool msk = h && (t < tx);
    float sg = msk ? sigma[m] : 0.f;
    float alpha = 1.f - __expf(-fmaxf(sg, 0.f) * STEP_F);
    float wgt = alpha * T;
    T *= (1.f - alpha + 1e-10f);
    if (msk) {
      float r0 = rgb[m*3+0], r1 = rgb[m*3+1], r2 = rgb[m*3+2];
      o0 += wgt / (1.f + __expf(-r0));
      o1 += wgt / (1.f + __expf(-r1));
      o2 += wgt / (1.f + __expf(-r2));
    }
    acc += wgt;
  }
  float bg = 1.f - acc;
  out[b*3+0] = o0 + bg;
  out[b*3+1] = o1 + bg;
  out[b*3+2] = o2 + bg;
}

// ---------------- launch ----------------
extern "C" void kernel_launch(void* const* d_in, const int* in_sizes, int n_in,
                              void* d_out, int out_size, void* d_ws, size_t ws_size,
                              hipStream_t stream) {
  const float* rays_o = (const float*)d_in[0];
  const float* rays_d = (const float*)d_in[1];
  const float* G1     = (const float*)d_in[2];
  const float* F      = (const float*)d_in[3];
  const float* Wd1    = (const float*)d_in[4];
  const float* bd1    = (const float*)d_in[5];
  const float* Wd2    = (const float*)d_in[6];
  const float* bd2    = (const float*)d_in[7];
  const float* Wc1    = (const float*)d_in[8];
  const float* bc1    = (const float*)d_in[9];
  const float* Wc2    = (const float*)d_in[10];
  const float* bc2    = (const float*)d_in[11];

  char* ws = (char*)d_ws;
  float4*    dirn    = (float4*)(ws + WS_DIRN);
  float*     tnear   = (float*)(ws + WS_TNEAR);
  float*     tmaxp   = (float*)(ws + WS_TMAX);
  int*       hit     = (int*)(ws + WS_HIT);
  _Float16*  fragb   = (_Float16*)(ws + WS_FRAG);
  float*     Ft      = (float*)(ws + WS_FT);
  float4*    G1t     = (float4*)(ws + WS_G1T);
  float*     sigmaws = (float*)(ws + WS_SIGMA);
  float*     rgbws   = (float*)(ws + WS_RGB);

  ray_setup_kernel<<<8, 256, 0, stream>>>(rays_o, rays_d, dirn, tnear, tmaxp, hit);
  transpose_G1_kernel<<<8192, 256, 0, stream>>>(G1, G1t);
  transpose_F_kernel<<<128, 256, 0, stream>>>(F, Ft);
  pack_weights_kernel<<<40, 256, 0, stream>>>(Wd1, Wc1, Wd2, Wc2, fragb);
  nerf_fused_kernel<<<4096, 128, 0, stream>>>(rays_o, dirn, tnear, G1t, (const float4*)Ft,
                                              fragb, bd1, bd2, bc1, bc2, sigmaws, rgbws);
  integrate_kernel<<<8, 256, 0, stream>>>(sigmaws, rgbws, tnear, tmaxp, hit, (float*)d_out);
}
